// HLGAN_26637387170379
// MI455X (gfx1250) — compile-verified, matrix operand order checked by reading, not run on hardware
//
#include <hip/hip_runtime.h>
#include <hip/hip_bf16.h>

// ---------------------------------------------------------------- constants
#define BATCH   64
#define IMAGE   224
#define PATCH   14
#define GRID_   16
#define NPATCH  256          // 16*16
#define NTOK    257          // NPATCH + 1
#define DMODEL  768
#define NHEADS  12
#define HDIM    64
#define QKVD    2304         // 3*DMODEL
#define KPATCH  224          // 196 padded to multiple of 32
#define NCLS    1000
#define NCLSP   1024         // padded to 64-col tile multiple

typedef __attribute__((ext_vector_type(16))) _Float16 v16h;
typedef __attribute__((ext_vector_type(8)))  _Float16 v8h;
typedef __attribute__((ext_vector_type(8)))  float    v8f;

// ---------------------------------------------------------------- helpers
// Convert f32 [R,C] -> f16 [Rpad,Cpad] with zero padding (row-major).
__global__ void vit_cvt_pad(const float* __restrict__ W, int R, int C,
                            _Float16* __restrict__ O, int Cpad, int total) {
    int idx = blockIdx.x * blockDim.x + threadIdx.x;
    if (idx >= total) return;
    int c = idx % Cpad, r = idx / Cpad;
    float v = (r < R && c < C) ? W[(size_t)r * C + c] : 0.0f;
    O[idx] = (_Float16)v;
}

// Patch matrix: P[b*256+p, c] = x[b, gy*14+py, gx*14+px], zero-padded c>=196.
__global__ void vit_patch_extract(const float* __restrict__ x,
                                  _Float16* __restrict__ P) {
    int idx = blockIdx.x * blockDim.x + threadIdx.x;   // 16384*224 exact
    int c = idx % KPATCH, row = idx / KPATCH;
    int p = row % NPATCH, b = row / NPATCH;
    _Float16 v = (_Float16)0.0f;
    if (c < PATCH * PATCH) {
        int py = c / PATCH, px = c % PATCH;
        int gy = p / GRID_, gx = p % GRID_;
        v = (_Float16)x[((size_t)b * IMAGE + gy * PATCH + py) * IMAGE + gx * PATCH + px];
    }
    P[idx] = v;
}

// tokens[b,i,:] = (i==0 ? cls : Ptok[b,i-1]) + pos[i]
__global__ void vit_assemble(const _Float16* __restrict__ Ptok,
                             const float* __restrict__ cls,
                             const float* __restrict__ pos,
                             _Float16* __restrict__ T0) {
    int idx = blockIdx.x * blockDim.x + threadIdx.x;   // B*NTOK*DMODEL exact
    int d = idx % DMODEL;
    int rest = idx / DMODEL;
    int i = rest % NTOK, b = rest / NTOK;
    float v = (i == 0) ? cls[d]
                       : (float)Ptok[((size_t)b * NPATCH + i - 1) * DMODEL + d];
    T0[idx] = (_Float16)(v + pos[(size_t)i * DMODEL + d]);
}

// ---------------------------------------------------------------- WMMA GEMM
// C[M,N] = A[M,K] * Bw[N,K]^T + bias.  One wave -> 16M x 64N tile.
// blockDim 128 (4 waves): mTile = bx*4+wave, nTiles = by*4 .. by*4+3.
// All M,N tile counts divisible by 4 by construction; store guarded by n_valid.
template <typename OutT>
__global__ void vit_gemm_wmma(const _Float16* __restrict__ A, int lda,
                              const _Float16* __restrict__ Bw, int ldb,
                              const float* __restrict__ bias,
                              OutT* __restrict__ C, int ldc,
                              int K, int n_valid) {
    const int lane  = threadIdx.x & 31;
    const int wave  = threadIdx.x >> 5;
    const int mTile = blockIdx.x * 4 + wave;
    const int nT0   = blockIdx.y * 4;

    const int l16 = lane & 15;               // M index for A frag, N index for B frag
    const int kb  = (lane >> 4) << 3;        // 0 or 8 (ISA A/B fragment K pattern)

    const _Float16* aRow = A  + (size_t)(mTile * 16 + l16) * lda + kb;
    const _Float16* bR0  = Bw + (size_t)((nT0 + 0) * 16 + l16) * ldb + kb;
    const _Float16* bR1  = Bw + (size_t)((nT0 + 1) * 16 + l16) * ldb + kb;
    const _Float16* bR2  = Bw + (size_t)((nT0 + 2) * 16 + l16) * ldb + kb;
    const _Float16* bR3  = Bw + (size_t)((nT0 + 3) * 16 + l16) * ldb + kb;

    v8f acc0 = {}, acc1 = {}, acc2 = {}, acc3 = {};

    for (int k0 = 0; k0 < K; k0 += 32) {
        // A fragment: two 16-byte chunks (K=[kb,kb+8) and K=[16+kb,16+kb+8))
        v8h a0 = *(const v8h*)(aRow + k0);
        v8h a1 = *(const v8h*)(aRow + k0 + 16);
        v16h a = __builtin_shufflevector(a0, a1, 0,1,2,3,4,5,6,7,8,9,10,11,12,13,14,15);

        v8h b00 = *(const v8h*)(bR0 + k0), b01 = *(const v8h*)(bR0 + k0 + 16);
        v8h b10 = *(const v8h*)(bR1 + k0), b11 = *(const v8h*)(bR1 + k0 + 16);
        v8h b20 = *(const v8h*)(bR2 + k0), b21 = *(const v8h*)(bR2 + k0 + 16);
        v8h b30 = *(const v8h*)(bR3 + k0), b31 = *(const v8h*)(bR3 + k0 + 16);
        v16h b0 = __builtin_shufflevector(b00, b01, 0,1,2,3,4,5,6,7,8,9,10,11,12,13,14,15);
        v16h b1 = __builtin_shufflevector(b10, b11, 0,1,2,3,4,5,6,7,8,9,10,11,12,13,14,15);
        v16h b2 = __builtin_shufflevector(b20, b21, 0,1,2,3,4,5,6,7,8,9,10,11,12,13,14,15);
        v16h b3 = __builtin_shufflevector(b30, b31, 0,1,2,3,4,5,6,7,8,9,10,11,12,13,14,15);

        acc0 = __builtin_amdgcn_wmma_f32_16x16x32_f16(false, a, false, b0, (short)0, acc0, false, false);
        acc1 = __builtin_amdgcn_wmma_f32_16x16x32_f16(false, a, false, b1, (short)0, acc1, false, false);
        acc2 = __builtin_amdgcn_wmma_f32_16x16x32_f16(false, a, false, b2, (short)0, acc2, false, false);
        acc3 = __builtin_amdgcn_wmma_f32_16x16x32_f16(false, a, false, b3, (short)0, acc3, false, false);
    }

    // C layout: VGPR r -> M = r + (lane<16?0:8); N = lane&15.
    const int mBase = mTile * 16 + ((lane >> 4) << 3);
    const int nBase = nT0 * 16 + l16;
#pragma unroll
    for (int r = 0; r < 8; ++r) {
        const size_t row = (size_t)(mBase + r) * ldc;
#pragma unroll
        for (int t = 0; t < 4; ++t) {
            int n = nBase + 16 * t;
            float v = (t == 0 ? acc0[r] : t == 1 ? acc1[r] : t == 2 ? acc2[r] : acc3[r]);
            if (n < n_valid) C[row + n] = (OutT)(v + bias[n]);
        }
    }
}

// ---------------------------------------------------------------- attention
// Local banded attention (radius 1): one thread per (b,h,i); <=3 keys.
__global__ void vit_attn_local(const _Float16* __restrict__ QKV,
                               _Float16* __restrict__ ATT) {
    int idx = blockIdx.x * blockDim.x + threadIdx.x;   // B*NHEADS*NTOK exact
    int i = idx % NTOK;
    int bh = idx / NTOK;
    int h = bh % NHEADS, b = bh / NHEADS;

    const _Float16* q = QKV + (size_t)(b * NTOK + i) * QKVD + h * HDIM;

    float s[3]; int valid[3];
    float mx = -3.0e38f;
#pragma unroll
    for (int t = 0; t < 3; ++t) {
        int j = i - 1 + t;
        valid[t] = (j >= 0 && j < NTOK);
        s[t] = -3.0e38f;
        if (valid[t]) {
            const _Float16* k = QKV + (size_t)(b * NTOK + j) * QKVD + DMODEL + h * HDIM;
            float d = 0.0f;
            for (int e = 0; e < HDIM; ++e) d += (float)q[e] * (float)k[e];
            s[t] = d * 0.125f;                       // 1/sqrt(64)
            mx = fmaxf(mx, s[t]);
        }
    }
    float e3[3], sum = 0.0f;
#pragma unroll
    for (int t = 0; t < 3; ++t) {
        e3[t] = valid[t] ? __expf(s[t] - mx) : 0.0f;
        sum += e3[t];
    }
    float inv = 1.0f / sum;

    const _Float16* v0 = QKV + (size_t)(b * NTOK + (i > 0 ? i - 1 : 0)) * QKVD + 2 * DMODEL + h * HDIM;
    const _Float16* v1 = QKV + (size_t)(b * NTOK + i) * QKVD + 2 * DMODEL + h * HDIM;
    const _Float16* v2 = QKV + (size_t)(b * NTOK + (i < NTOK - 1 ? i + 1 : i)) * QKVD + 2 * DMODEL + h * HDIM;
    _Float16* out = ATT + (size_t)(b * NTOK + i) * DMODEL + h * HDIM;
    for (int d = 0; d < HDIM; ++d) {
        float o = e3[0] * (float)v0[d] + e3[1] * (float)v1[d] + e3[2] * (float)v2[d];
        out[d] = (_Float16)(o * inv);
    }
}

// Global attention: one wave per (b,h,i). Pass1: scores->LDS + softmax stats.
// Pass2: out[d] = sum_j p[j]*v[j][d], d = lane, lane+32.
__global__ void vit_attn_global(const _Float16* __restrict__ QKV,
                                _Float16* __restrict__ ATT) {
    __shared__ float sc[8][NTOK + 7];
    const int lane = threadIdx.x & 31;
    const int wave = threadIdx.x >> 5;
    const int idx = blockIdx.x * 8 + wave;             // B*NHEADS*NTOK exact
    const int i = idx % NTOK;
    const int bh = idx / NTOK;
    const int h = bh % NHEADS, b = bh / NHEADS;

    const _Float16* qp = QKV + (size_t)(b * NTOK + i) * QKVD + h * HDIM;
    v8h qh[8];
#pragma unroll
    for (int t = 0; t < 8; ++t) qh[t] = *(const v8h*)(qp + 8 * t);

    // pass 1: per-lane strided keys
    float mx = -3.0e38f;
    for (int j = lane; j < NTOK; j += 32) {
        const _Float16* kp = QKV + (size_t)(b * NTOK + j) * QKVD + DMODEL + h * HDIM;
        float d = 0.0f;
#pragma unroll
        for (int t = 0; t < 8; ++t) {
            v8h kh = *(const v8h*)(kp + 8 * t);
#pragma unroll
            for (int e = 0; e < 8; ++e) d += (float)qh[t][e] * (float)kh[e];
        }
        d *= 0.125f;
        sc[wave][j] = d;
        mx = fmaxf(mx, d);
    }
#pragma unroll
    for (int off = 16; off; off >>= 1) mx = fmaxf(mx, __shfl_xor(mx, off, 32));

    float sum = 0.0f;
    for (int j = lane; j < NTOK; j += 32) {
        float e = __expf(sc[wave][j] - mx);
        sc[wave][j] = e;
        sum += e;
    }
#pragma unroll
    for (int off = 16; off; off >>= 1) sum += __shfl_xor(sum, off, 32);
    const float inv = 1.0f / sum;
    __syncthreads();

    // pass 2: weighted sum over values
    _Float16* out = ATT + (size_t)(b * NTOK + i) * DMODEL + h * HDIM;
    const _Float16* vbase = QKV + (size_t)b * NTOK * QKVD + 2 * DMODEL + h * HDIM;
#pragma unroll
    for (int t = 0; t < 2; ++t) {
        const int d = lane + 32 * t;
        float o = 0.0f;
        for (int j = 0; j < NTOK; ++j)
            o += sc[wave][j] * (float)vbase[(size_t)j * QKVD + d];
        out[d] = (_Float16)(o * inv);
    }
}

// ---------------------------------------------------------------- layernorm
__global__ void vit_layernorm(const _Float16* __restrict__ X,
                              const float* __restrict__ g,
                              const float* __restrict__ be,
                              _Float16* __restrict__ Y) {
    __shared__ float red[256];
    const int row = blockIdx.x;
    const int t = threadIdx.x;
    const _Float16* xr = X + (size_t)row * DMODEL;

    float x0 = (float)xr[t], x1 = (float)xr[t + 256], x2 = (float)xr[t + 512];
    red[t] = x0 + x1 + x2;
    __syncthreads();
    for (int s = 128; s; s >>= 1) { if (t < s) red[t] += red[t + s]; __syncthreads(); }
    const float mu = red[0] * (1.0f / DMODEL);
    __syncthreads();

    float d0 = x0 - mu, d1 = x1 - mu, d2 = x2 - mu;
    red[t] = d0 * d0 + d1 * d1 + d2 * d2;
    __syncthreads();
    for (int s = 128; s; s >>= 1) { if (t < s) red[t] += red[t + s]; __syncthreads(); }
    const float rs = rsqrtf(red[0] * (1.0f / DMODEL) + 1e-5f);

    _Float16* yr = Y + (size_t)row * DMODEL;
    yr[t]       = (_Float16)(d0 * rs * g[t]       + be[t]);
    yr[t + 256] = (_Float16)(d1 * rs * g[t + 256] + be[t + 256]);
    yr[t + 512] = (_Float16)(d2 * rs * g[t + 512] + be[t + 512]);
}

// ---------------------------------------------------------------- launcher
extern "C" void kernel_launch(void* const* d_in, const int* in_sizes, int n_in,
                              void* d_out, int out_size, void* d_ws, size_t ws_size,
                              hipStream_t stream) {
    const float* x        = (const float*)d_in[0];
    const float* w_patch  = (const float*)d_in[1];
    const float* b_patch  = (const float*)d_in[2];
    const float* cls_tok  = (const float*)d_in[3];
    const float* pos_emb  = (const float*)d_in[4];
    const float* wqkv_l   = (const float*)d_in[5];
    const float* bqkv_l   = (const float*)d_in[6];
    const float* wo_l     = (const float*)d_in[7];
    const float* bo_l     = (const float*)d_in[8];
    const float* wqkv_g   = (const float*)d_in[9];
    const float* bqkv_g   = (const float*)d_in[10];
    const float* wo_g     = (const float*)d_in[11];
    const float* bo_g     = (const float*)d_in[12];
    const float* g1       = (const float*)d_in[13];
    const float* be1      = (const float*)d_in[14];
    const float* g2       = (const float*)d_in[15];
    const float* be2      = (const float*)d_in[16];
    const float* w_cls    = (const float*)d_in[17];
    const float* b_cls    = (const float*)d_in[18];
    float* out            = (float*)d_out;

    // -------- workspace carve (all f16 activations/weights)
    char* p = (char*)d_ws;
    auto carve = [&](size_t bytes) {
        char* r = p;
        p += (bytes + 255) & ~(size_t)255;
        return r;
    };
    _Float16* wPatchH = (_Float16*)carve((size_t)DMODEL * KPATCH * 2);
    _Float16* wQkvLH  = (_Float16*)carve((size_t)QKVD * DMODEL * 2);
    _Float16* wOLH    = (_Float16*)carve((size_t)DMODEL * DMODEL * 2);
    _Float16* wQkvGH  = (_Float16*)carve((size_t)QKVD * DMODEL * 2);
    _Float16* wOGH    = (_Float16*)carve((size_t)DMODEL * DMODEL * 2);
    _Float16* wClsH   = (_Float16*)carve((size_t)NCLSP * DMODEL * 2);
    _Float16* T0      = (_Float16*)carve((size_t)BATCH * NTOK * DMODEL * 2);
    _Float16* QKV     = (_Float16*)carve((size_t)BATCH * NTOK * QKVD * 2);
    _Float16* ATT     = (_Float16*)carve((size_t)BATCH * NTOK * DMODEL * 2);
    _Float16* T1      = (_Float16*)carve((size_t)BATCH * NTOK * DMODEL * 2);
    _Float16* Pmat    = (_Float16*)T1;    // alias: patch stage only
    _Float16* Ptok    = (_Float16*)ATT;   // alias: patch stage only

    const int MROWS = BATCH * NTOK;       // 16448 = 16*1028, 1028 = 4*257
    auto cdiv = [](int a, int b) { return (a + b - 1) / b; };

    // -------- 1. weight conversion (f32 -> f16, padded)
    {
        int tot;
        tot = DMODEL * KPATCH;
        vit_cvt_pad<<<cdiv(tot, 256), 256, 0, stream>>>(w_patch, DMODEL, PATCH * PATCH, wPatchH, KPATCH, tot);
        tot = QKVD * DMODEL;
        vit_cvt_pad<<<cdiv(tot, 256), 256, 0, stream>>>(wqkv_l, QKVD, DMODEL, wQkvLH, DMODEL, tot);
        vit_cvt_pad<<<cdiv(tot, 256), 256, 0, stream>>>(wqkv_g, QKVD, DMODEL, wQkvGH, DMODEL, tot);
        tot = DMODEL * DMODEL;
        vit_cvt_pad<<<cdiv(tot, 256), 256, 0, stream>>>(wo_l, DMODEL, DMODEL, wOLH, DMODEL, tot);
        vit_cvt_pad<<<cdiv(tot, 256), 256, 0, stream>>>(wo_g, DMODEL, DMODEL, wOGH, DMODEL, tot);
        tot = NCLSP * DMODEL;
        vit_cvt_pad<<<cdiv(tot, 256), 256, 0, stream>>>(w_cls, NCLS, DMODEL, wClsH, DMODEL, tot);
    }

    // -------- 2. patch extraction + patch-embed GEMM + token assembly
    vit_patch_extract<<<(BATCH * NPATCH * KPATCH) / 256, 256, 0, stream>>>(x, Pmat);
    {   // Ptok[16384,768] = Pmat[16384,224] @ wPatchH[768,224]^T + b_patch
        dim3 g((BATCH * NPATCH / 16) / 4, DMODEL / 64);
        vit_gemm_wmma<_Float16><<<g, 128, 0, stream>>>(Pmat, KPATCH, wPatchH, KPATCH,
                                                       b_patch, Ptok, DMODEL, KPATCH, DMODEL);
    }
    vit_assemble<<<(BATCH * NTOK * DMODEL) / 256, 256, 0, stream>>>(Ptok, cls_tok, pos_emb, T0);

    // -------- 3. local block: QKV -> banded attention -> proj -> LN
    {
        dim3 g((MROWS / 16) / 4, QKVD / 64);
        vit_gemm_wmma<_Float16><<<g, 128, 0, stream>>>(T0, DMODEL, wQkvLH, DMODEL,
                                                       bqkv_l, QKV, QKVD, DMODEL, QKVD);
    }
    vit_attn_local<<<(BATCH * NHEADS * NTOK) / 256, 256, 0, stream>>>(QKV, ATT);
    {
        dim3 g((MROWS / 16) / 4, DMODEL / 64);
        vit_gemm_wmma<_Float16><<<g, 128, 0, stream>>>(ATT, DMODEL, wOLH, DMODEL,
                                                       bo_l, T1, DMODEL, DMODEL, DMODEL);
    }
    vit_layernorm<<<MROWS, 256, 0, stream>>>(T1, g1, be1, T0);

    // -------- 4. global block: QKV -> full attention -> proj -> LN
    {
        dim3 g((MROWS / 16) / 4, QKVD / 64);
        vit_gemm_wmma<_Float16><<<g, 128, 0, stream>>>(T0, DMODEL, wQkvGH, DMODEL,
                                                       bqkv_g, QKV, QKVD, DMODEL, QKVD);
    }
    vit_attn_global<<<(BATCH * NHEADS * NTOK) / 8, 256, 0, stream>>>(QKV, ATT);
    {
        dim3 g((MROWS / 16) / 4, DMODEL / 64);
        vit_gemm_wmma<_Float16><<<g, 128, 0, stream>>>(ATT, DMODEL, wOGH, DMODEL,
                                                       bo_g, T1, DMODEL, DMODEL, DMODEL);
    }
    vit_layernorm<<<MROWS, 256, 0, stream>>>(T1, g2, be2, T0);

    // -------- 5. classifier: pooled = T0[:,0,:] read with lda = NTOK*DMODEL
    {
        dim3 g((BATCH / 16) / 4, NCLSP / 64);   // (1, 16)
        vit_gemm_wmma<float><<<g, 128, 0, stream>>>(T0, NTOK * DMODEL, wClsH, DMODEL,
                                                    b_cls, out, NCLS, DMODEL, NCLS);
    }
}
